// VectorQuant_36593121362126
// MI455X (gfx1250) — matrix operand order repaired and checked by reading, not compiled, not run on hardware
//
#include <hip/hip_runtime.h>
#include <hip/hip_bf16.h>
#include <math.h>

typedef __attribute__((ext_vector_type(16))) _Float16 v16h;
typedef __attribute__((ext_vector_type(4)))  _Float16 v4h;
typedef __attribute__((ext_vector_type(8)))  float    v8f;

#define N_ 8
#define S_ 2048
#define C_ 4
#define V_ 128
#define K_ 512
#define MROWS (N_*S_)          // 16384 rows per c
#define MT 2                   // M-tiles (16 rows each) per wave

// d_out layout (floats, concatenated in reference return order)
#define OFF_DISC 0
#define OFF_IDX  (N_*S_*C_*V_)              // 8388608
#define OFF_VQ   (OFF_IDX + N_*S_*C_)       // 8454144
#define OFF_ENC  (OFF_VQ  + N_*S_*C_)       // 8519680
#define OFF_ENT  (OFF_ENC + N_*S_*C_)       // 8585216

// ---------------------------------------------------------------------------
// Kernel A: normalize embeddings; produce f32 copy, e2, and split hi/lo f16.
// One wave per (c,k) row of 128 elements; lane handles 4 contiguous v.
// ---------------------------------------------------------------------------
__global__ void __launch_bounds__(256) vq_prep_emb(
    const float* __restrict__ e0,
    float* __restrict__ emb_f32,
    _Float16* __restrict__ emb_hi,
    _Float16* __restrict__ emb_lo,
    float* __restrict__ e2out)
{
  const float tn = 0.06f * sqrtf((float)V_);
  const int lane = threadIdx.x & 31;
  const int wave = threadIdx.x >> 5;
  const int row  = blockIdx.x * 8 + wave;          // 0 .. C_*K_-1
  const float* src = e0 + (size_t)row * V_ + lane * 4;
  float4 v = *(const float4*)src;

  float ss = v.x*v.x + v.y*v.y + v.z*v.z + v.w*v.w;
  #pragma unroll
  for (int m = 16; m >= 1; m >>= 1) ss += __shfl_xor(ss, m, 32);
  const float nrm = sqrtf(ss);

  float4 ef;
  ef.x = tn * v.x / nrm;
  ef.y = tn * v.y / nrm;
  ef.z = tn * v.z / nrm;
  ef.w = tn * v.w / nrm;
  *(float4*)(emb_f32 + (size_t)row * V_ + lane * 4) = ef;

  float s2 = ef.x*ef.x + ef.y*ef.y + ef.z*ef.z + ef.w*ef.w;
  #pragma unroll
  for (int m = 16; m >= 1; m >>= 1) s2 += __shfl_xor(s2, m, 32);
  if (lane == 0) e2out[row] = s2;

  v4h hi, lo;
  {
    _Float16 h0 = (_Float16)ef.x; hi[0] = h0; lo[0] = (_Float16)(ef.x - (float)h0);
    _Float16 h1 = (_Float16)ef.y; hi[1] = h1; lo[1] = (_Float16)(ef.y - (float)h1);
    _Float16 h2 = (_Float16)ef.z; hi[2] = h2; lo[2] = (_Float16)(ef.z - (float)h2);
    _Float16 h3 = (_Float16)ef.w; hi[3] = h3; lo[3] = (_Float16)(ef.w - (float)h3);
  }
  *(v4h*)(emb_hi + (size_t)row * V_ + lane * 4) = hi;
  *(v4h*)(emb_lo + (size_t)row * V_ + lane * 4) = lo;
}

// helper for epilogue elementwise math
__device__ __forceinline__ void vq_elem(float x0v, float ev, float tn, float nrm,
                                        float& vq, float& ee, float& dout)
{
  float xn = tn * x0v / nrm;       // normalized x (matches reference elementwise op)
  float d  = xn - ev;  vq += d * d;
  float dx = xn - x0v; ee += dx * dx;
  dout = xn + (ev - xn);           // discrete = x + (output - x), mimic fp order
}

// ---------------------------------------------------------------------------
// Kernel B: main GEMM (split-f16 WMMA) + argmin + outputs.
// Wave handles MT=2 consecutive 16-row M-tiles for one c.
// Lane mapping (wave32): m16 = lane&15 (row within tile), h = lane>>4.
// A-frag (16-bit A 16x32): a[j] = x[m16][kbase + (j>>3)*16 + h*8 + (j&7)]
// B-frag (32x16): b[j] = e[col = nt*16 + m16][kbase + (j>>3)*16 + h*8 + (j&7)]
// C/D: c[r] = D[r + 8h][m16]
// ---------------------------------------------------------------------------
__global__ void __launch_bounds__(256) vq_main(
    const float* __restrict__ x0,
    const float* __restrict__ emb_f32,
    const _Float16* __restrict__ emb_hi,
    const _Float16* __restrict__ emb_lo,
    const float* __restrict__ e2buf,
    float* __restrict__ out)
{
  const float tn = 0.06f * sqrtf((float)V_);
  const int lane = threadIdx.x & 31;
  const int gw   = blockIdx.x * 8 + (threadIdx.x >> 5);   // 0..2047
  const int c    = gw >> 9;                               // 512 waves per c
  const int tg   = gw & 511;                              // tile group within c
  const int m16  = lane & 15;
  const int h    = lane >> 4;

  v16h  a_hi[MT][4], a_lo[MT][4];
  float x2r[MT][8];
  float norm0[MT];

  // ---- Phase 1: load x rows, normalize, build split A-fragments -----------
  #pragma unroll
  for (int mt = 0; mt < MT; ++mt) {
    const int T    = tg * MT + mt;            // tile index 0..1023
    const int grow = T * 16 + m16;            // global row (n*S+s)
    const float* xr = x0 + ((size_t)grow * C_ + c) * V_;

    float xa[64];
    float ss = 0.f;
    #pragma unroll
    for (int b = 0; b < 8; ++b) {
      const int v0 = b * 16 + h * 8;
      float4 t0 = *(const float4*)(xr + v0);
      float4 t1 = *(const float4*)(xr + v0 + 4);
      xa[b*8+0]=t0.x; xa[b*8+1]=t0.y; xa[b*8+2]=t0.z; xa[b*8+3]=t0.w;
      xa[b*8+4]=t1.x; xa[b*8+5]=t1.y; xa[b*8+6]=t1.z; xa[b*8+7]=t1.w;
      ss += t0.x*t0.x + t0.y*t0.y + t0.z*t0.z + t0.w*t0.w;
      ss += t1.x*t1.x + t1.y*t1.y + t1.z*t1.z + t1.w*t1.w;
    }
    ss += __shfl_xor(ss, 16, 32);             // full-row sum (row split lane m / m+16)
    const float nrm = sqrtf(ss);
    norm0[mt] = nrm;

    float x2p = 0.f;
    #pragma unroll
    for (int i = 0; i < 64; ++i) {
      float xn = tn * xa[i] / nrm;
      xa[i] = xn;
      x2p += xn * xn;
    }
    float x2tot = x2p + __shfl_xor(x2p, 16, 32);
    #pragma unroll
    for (int r = 0; r < 8; ++r)
      x2r[mt][r] = __shfl(x2tot, r + h * 8, 32);   // x2 of row r+8h

    #pragma unroll
    for (int kc = 0; kc < 4; ++kc) {
      v16h hi, lo;
      #pragma unroll
      for (int j = 0; j < 16; ++j) {
        const int g = j >> 3, t = j & 7;
        float xv = xa[(2*kc + g) * 8 + t];    // K = kc*32 + g*16 + h*8 + t
        _Float16 hv = (_Float16)xv;
        hi[j] = hv;
        lo[j] = (_Float16)(xv - (float)hv);
      }
      a_hi[mt][kc] = hi;
      a_lo[mt][kc] = lo;
    }
  }

  // ---- Phase 2: sweep all 512 codes via WMMA, track argmin of d2 ----------
  float bestd[MT][8];
  int   bestk[MT][8];
  #pragma unroll
  for (int mt = 0; mt < MT; ++mt)
    #pragma unroll
    for (int r = 0; r < 8; ++r) { bestd[mt][r] = __builtin_inff(); bestk[mt][r] = 0x7FFFFFFF; }

  const _Float16* ehc = emb_hi + (size_t)c * K_ * V_;
  const _Float16* elc = emb_lo + (size_t)c * K_ * V_;

  for (int nt = 0; nt < 32; ++nt) {
    const int col = nt * 16 + m16;
    const float e2v = e2buf[c * K_ + col];

    v16h bh[4], bl[4];
    #pragma unroll
    for (int kc = 0; kc < 4; ++kc) {
      typedef __attribute__((ext_vector_type(8))) _Float16 v8h_;
      const size_t off = (size_t)col * V_ + kc * 32 + h * 8;
      bh[kc] = __builtin_shufflevector(*(const v8h_*)(ehc + off), *(const v8h_*)(ehc + off + 16), 0,1,2,3,4,5,6,7,8,9,10,11,12,13,14,15);
      bl[kc] = __builtin_shufflevector(*(const v8h_*)(elc + off), *(const v8h_*)(elc + off + 16), 0,1,2,3,4,5,6,7,8,9,10,11,12,13,14,15);
    }

    #pragma unroll
    for (int mt = 0; mt < MT; ++mt) {
      v8f acc = {};
      #pragma unroll
      for (int kc = 0; kc < 4; ++kc) {
        acc = __builtin_amdgcn_wmma_f32_16x16x32_f16(false, a_hi[mt][kc], false, bh[kc], (short)0, acc, false, false);
        acc = __builtin_amdgcn_wmma_f32_16x16x32_f16(false, a_lo[mt][kc], false, bh[kc], (short)0, acc, false, false);
        acc = __builtin_amdgcn_wmma_f32_16x16x32_f16(false, a_hi[mt][kc], false, bl[kc], (short)0, acc, false, false);
      }
      #pragma unroll
      for (int r = 0; r < 8; ++r) {
        float d2 = (x2r[mt][r] + e2v) - 2.0f * acc[r];
        // branchless: avoid short-circuit ||/&& so this lowers to v_cmp + v_cndmask
        int better = (int)(d2 < bestd[mt][r]) |
                     ((int)(d2 == bestd[mt][r]) & (int)(col < bestk[mt][r]));
        bestd[mt][r] = better ? d2  : bestd[mt][r];
        bestk[mt][r] = better ? col : bestk[mt][r];
      }
    }
  }

  // ---- Reduce argmin across the 16 lanes of each half (rows r+8h) ---------
  #pragma unroll
  for (int mt = 0; mt < MT; ++mt) {
    #pragma unroll
    for (int r = 0; r < 8; ++r) {
      float d = bestd[mt][r];
      int   k = bestk[mt][r];
      #pragma unroll
      for (int m = 1; m <= 8; m <<= 1) {
        float od = __shfl_xor(d, m, 32);
        int   ok = __shfl_xor(k, m, 32);
        int take = (int)(od < d) | ((int)(od == d) & (int)(ok < k));
        d = take ? od : d;
        k = take ? ok : k;
      }
      bestd[mt][r] = d;
      bestk[mt][r] = k;
    }
  }

  // ---- Epilogue: gather selected codes, penalties, write outputs ----------
  #pragma unroll
  for (int mt = 0; mt < MT; ++mt) {
    // best-k for this lane's own data row m16 (may live in the other half)
    int own = bestk[mt][0];
    #pragma unroll
    for (int r = 1; r < 8; ++r)
      own = ((m16 & 7) == r) ? bestk[mt][r] : own;   // cndmask chain
    int other = __shfl_xor(own, 16, 32);
    const int bk = ((m16 >> 3) == h) ? own : other;

    const int T    = tg * MT + mt;
    const int grow = T * 16 + m16;
    const float* xr = x0      + ((size_t)grow * C_ + c) * V_;
    const float* er = emb_f32 + ((size_t)c * K_ + bk) * V_;
    float*       dr = out + OFF_DISC + ((size_t)grow * C_ + c) * V_;
    const float  nrm = norm0[mt];

    float vq = 0.f, ee = 0.f;
    #pragma unroll
    for (int b = 0; b < 8; ++b) {
      const int v0 = b * 16 + h * 8;
      float4 xA = *(const float4*)(xr + v0);
      float4 xB = *(const float4*)(xr + v0 + 4);
      float4 eA = *(const float4*)(er + v0);
      float4 eB = *(const float4*)(er + v0 + 4);
      float4 oA, oB;
      vq_elem(xA.x, eA.x, tn, nrm, vq, ee, oA.x);
      vq_elem(xA.y, eA.y, tn, nrm, vq, ee, oA.y);
      vq_elem(xA.z, eA.z, tn, nrm, vq, ee, oA.z);
      vq_elem(xA.w, eA.w, tn, nrm, vq, ee, oA.w);
      vq_elem(xB.x, eB.x, tn, nrm, vq, ee, oB.x);
      vq_elem(xB.y, eB.y, tn, nrm, vq, ee, oB.y);
      vq_elem(xB.z, eB.z, tn, nrm, vq, ee, oB.z);
      vq_elem(xB.w, eB.w, tn, nrm, vq, ee, oB.w);
      *(float4*)(dr + v0)     = oA;
      *(float4*)(dr + v0 + 4) = oB;
    }
    vq += __shfl_xor(vq, 16, 32);
    ee += __shfl_xor(ee, 16, 32);

    if (lane < 16) {                           // lanes 0..15 own rows 0..15
      const size_t si = (size_t)grow * C_ + c;
      out[OFF_IDX + si] = (float)(bk + c * K_);
      out[OFF_VQ  + si] = vq;
      out[OFF_ENC + si] = vq + ee;
    }
  }
}

// ---------------------------------------------------------------------------
// Kernel C: histogram over K bins (index = index1 mod K) + entropy.
// ---------------------------------------------------------------------------
__global__ void __launch_bounds__(512) vq_entropy(
    const float* __restrict__ idx1, float* __restrict__ ent)
{
  __shared__ int   hist[K_];
  __shared__ float red[512];
  const int tid = threadIdx.x;
  hist[tid] = 0;
  __syncthreads();
  for (int i = tid; i < N_ * S_ * C_; i += 512) {
    int k = ((int)idx1[i]) & (K_ - 1);
    atomicAdd(&hist[k], 1);
  }
  __syncthreads();
  int hv = hist[tid];
  float p = (float)hv / (float)(N_ * S_);
  red[tid] = (hv > 0) ? -(p * logf(p)) : 0.f;
  __syncthreads();
  for (int s = 256; s > 0; s >>= 1) {
    if (tid < s) red[tid] += red[tid + s];
    __syncthreads();
  }
  if (tid == 0) *ent = red[0];
}

// ---------------------------------------------------------------------------
extern "C" void kernel_launch(void* const* d_in, const int* in_sizes, int n_in,
                              void* d_out, int out_size, void* d_ws, size_t ws_size,
                              hipStream_t stream)
{
  const float* x0 = (const float*)d_in[0];   // (N,S,C,V) f32
  const float* e0 = (const float*)d_in[1];   // (C,K,V)   f32
  float* out = (float*)d_out;

  char* ws = (char*)d_ws;
  float*    emb_f32 = (float*)ws;                                   // 1 MB
  _Float16* emb_hi  = (_Float16*)(ws + (size_t)C_ * K_ * V_ * 4);   // 512 KB
  _Float16* emb_lo  = emb_hi + (size_t)C_ * K_ * V_;                // 512 KB
  float*    e2      = (float*)(emb_lo + (size_t)C_ * K_ * V_);      // 8 KB

  vq_prep_emb<<<256, 256, 0, stream>>>(e0, emb_f32, emb_hi, emb_lo, e2);
  vq_main   <<<256, 256, 0, stream>>>(x0, emb_f32, emb_hi, emb_lo, e2, out);
  vq_entropy<<<1,   512, 0, stream>>>(out + OFF_IDX, out + OFF_ENT);
}
